// MSANet_4887672782928
// MI455X (gfx1250) — compile-verified, matrix operand order checked by reading, not run on hardware
//
#include <hip/hip_runtime.h>

#define NB 16
#define NC 512
#define NH 32
#define NW 32
#define NL 1024      // H*W
#define NHEADS 4
#define NDH 128
#define NINNER 512
#define KSTR 136     // padded K-tile row stride (136*2B = 272B: 16B-aligned, bank step 4)
#define VSTR 72      // padded V^T-tile row stride (144B: 16B-aligned, bank step 36)

typedef __bf16 bf16;
typedef __attribute__((ext_vector_type(16))) __bf16 v16bf;
typedef __attribute__((ext_vector_type(8)))  __bf16 bf16x8;
typedef __attribute__((ext_vector_type(4)))  __bf16 bf16x4;
typedef __attribute__((ext_vector_type(8)))  float  v8f;

union Frag16 { v16bf v; bf16x8 h[2]; };

__device__ __forceinline__ v8f wmma_bf16(const Frag16& a, const Frag16& b, v8f c) {
  // D = A(16x32 bf16) * B(32x16 bf16) + C(16x16 f32)
  return __builtin_amdgcn_wmma_f32_16x16x32_bf16(false, a.v, false, b.v, (short)0, c,
                                                 false, false);
}

// Async global -> LDS b128 copy (GV mode). ldsOff = byte offset within LDS
// (low 32 bits of a generic __shared__ pointer), ga = 64-bit global address.
__device__ __forceinline__ void async_ld_b128(unsigned ldsOff, const void* gptr) {
  unsigned long long ga = (unsigned long long)(uintptr_t)gptr;
  asm volatile("global_load_async_to_lds_b128 %0, %1, off"
               :: "v"(ldsOff), "v"(ga) : "memory");
}
__device__ __forceinline__ void wait_async0() {
  asm volatile("s_wait_asynccnt 0x0" ::: "memory");
}

// ---------------------------------------------------------------------------
// Kernel 1: per-(b,c) mean / inv-std over 1024 spatial elements (ddof = 1).
// ---------------------------------------------------------------------------
__global__ __launch_bounds__(128) void stats_kernel(const float* __restrict__ content,
                                                    const float* __restrict__ style,
                                                    float* __restrict__ stats) {
  const int idx = blockIdx.x;
  const bool isStyle = idx >= NB * NC;
  const int bc = isStyle ? idx - NB * NC : idx;
  const float* src = (isStyle ? style : content) + (size_t)bc * NL;
  float s = 0.f, ss = 0.f;
  for (int i = threadIdx.x; i < NL; i += 128) {
    float x = src[i];
    s += x; ss += x * x;
  }
#pragma unroll
  for (int o = 16; o > 0; o >>= 1) {
    s  += __shfl_xor(s,  o, 32);
    ss += __shfl_xor(ss, o, 32);
  }
  __shared__ float red[2][4];
  const int wave = threadIdx.x >> 5;
  if ((threadIdx.x & 31) == 0) { red[0][wave] = s; red[1][wave] = ss; }
  __syncthreads();
  if (threadIdx.x == 0) {
    s  = red[0][0] + red[0][1] + red[0][2] + red[0][3];
    ss = red[1][0] + red[1][1] + red[1][2] + red[1][3];
    float mean = s * (1.0f / NL);
    float var  = (ss - (float)NL * mean * mean) * (1.0f / (NL - 1)) + 1e-5f;
    stats[idx * 2 + 0] = mean;
    stats[idx * 2 + 1] = rsqrtf(var);
  }
}

// ---------------------------------------------------------------------------
// Kernel 2: QKV projection GEMM. out[l,o] = sum_c Xn[l,c] * W[o,c]
//   z%3: 0=q (norm content, *SCALE, [b,h,l,d]),
//        1=k (norm style + pos emb folded,  [b,h,l,d]),
//        2=v (raw style, stored TRANSPOSED [b,h,d,l] for async staging later).
// Branch-free normalization: always load (mean, inv), select (0,1) for v.
// ---------------------------------------------------------------------------
__global__ __launch_bounds__(128) void qkv_kernel(
    const float* __restrict__ content, const float* __restrict__ style,
    const float* __restrict__ Wq, const float* __restrict__ Wk,
    const float* __restrict__ Wv, const float* __restrict__ pos_h,
    const float* __restrict__ pos_w, const float* __restrict__ stats,
    bf16* __restrict__ qo, bf16* __restrict__ ko, bf16* __restrict__ vo) {
  __shared__ bf16 Xs[64 * 32];   // [l][c] bf16, K-contiguous for A operand
  __shared__ bf16 Ws[64 * 32];   // [o][c] bf16, K-contiguous for B operand
  const int b     = blockIdx.z / 3;
  const int which = blockIdx.z % 3;
  const float* X  = (which == 0) ? content : style;
  const float* Wm = (which == 0) ? Wq : (which == 1) ? Wk : Wv;
  bf16* dst       = (which == 0) ? qo : (which == 1) ? ko : vo;
  const float* st = stats + (which == 0 ? 0 : 2 * NB * NC);
  const bool donorm = (which < 2);
  const int l0 = blockIdx.y * 64;
  const int o0 = blockIdx.x * 64;
  const int tid = threadIdx.x;
  const int lane = tid & 31;
  const int wave = tid >> 5;
  const int kb   = (lane & 16) ? 8 : 0;
  const int mrow = lane & 15;

  v8f acc[4] = {};
  for (int kc = 0; kc < NC; kc += 32) {
    // ---- issue all global loads first (MLP), then convert/store ----
    float4 x4[4], w4[4];
    float2 ms[4];
#pragma unroll
    for (int it = 0; it < 4; ++it) {
      int e  = it * 512 + tid * 4;
      int cl = e >> 6, ll = e & 63;
      int c  = kc + cl;
      x4[it] = *(const float4*)(X + ((size_t)b * NC + c) * NL + l0 + ll);
      ms[it] = *(const float2*)(st + (b * NC + c) * 2);  // always valid memory
    }
#pragma unroll
    for (int it = 0; it < 4; ++it) {
      int e  = it * 512 + tid * 4;
      int ol = e >> 5, cl = e & 31;
      w4[it] = *(const float4*)(Wm + (size_t)(o0 + ol) * NC + kc + cl);
    }
#pragma unroll
    for (int it = 0; it < 4; ++it) {
      int e  = it * 512 + tid * 4;
      int cl = e >> 6, ll = e & 63;
      float mean = donorm ? ms[it].x : 0.0f;   // branch-free select
      float inv  = donorm ? ms[it].y : 1.0f;
      float4 xv = x4[it];
      xv.x = (xv.x - mean) * inv; xv.y = (xv.y - mean) * inv;
      xv.z = (xv.z - mean) * inv; xv.w = (xv.w - mean) * inv;
      Xs[(ll + 0) * 32 + cl] = (bf16)xv.x;
      Xs[(ll + 1) * 32 + cl] = (bf16)xv.y;
      Xs[(ll + 2) * 32 + cl] = (bf16)xv.z;
      Xs[(ll + 3) * 32 + cl] = (bf16)xv.w;
    }
#pragma unroll
    for (int it = 0; it < 4; ++it) {
      int e  = it * 512 + tid * 4;
      int ol = e >> 5, cl = e & 31;
      bf16x4 p;
      p[0] = (bf16)w4[it].x; p[1] = (bf16)w4[it].y;
      p[2] = (bf16)w4[it].z; p[3] = (bf16)w4[it].w;
      *(bf16x4*)&Ws[ol * 32 + cl] = p;
    }
    __syncthreads();
    Frag16 a;
    const bf16* xr = &Xs[(wave * 16 + mrow) * 32];
    a.h[0] = *(const bf16x8*)(xr + kb);
    a.h[1] = *(const bf16x8*)(xr + kb + 16);
#pragma unroll
    for (int ot = 0; ot < 4; ++ot) {
      Frag16 bf;
      const bf16* wr = &Ws[(ot * 16 + mrow) * 32];
      bf.h[0] = *(const bf16x8*)(wr + kb);
      bf.h[1] = *(const bf16x8*)(wr + kb + 16);
      acc[ot] = wmma_bf16(a, bf, acc[ot]);
    }
    __syncthreads();
  }
  // epilogue
  const float scale = 0.088388347648318447f;   // 128^-0.5
#pragma unroll
  for (int ot = 0; ot < 4; ++ot) {
#pragma unroll
    for (int r = 0; r < 8; ++r) {
      int l = l0 + wave * 16 + r + ((lane & 16) ? 8 : 0);
      int o = o0 + ot * 16 + mrow;
      int h = o >> 7, d = o & 127;
      float val = acc[ot][r];
      size_t bh = (size_t)b * NHEADS + h;
      if (which == 0) {
        val *= scale;                                  // q pre-scaled
        dst[(bh * NL + l) * NDH + d] = (bf16)val;
      } else if (which == 1) {                         // fold emb into k
        int x = l >> 5, y = l & 31;
        val += pos_h[x * NDH + d] + pos_w[y * NDH + d];
        dst[(bh * NL + l) * NDH + d] = (bf16)val;
      } else {                                         // v transposed [bh,d,l]
        dst[(bh * NDH + d) * NL + l] = (bf16)val;
      }
    }
  }
}

// ---------------------------------------------------------------------------
// Kernel 3: flash-style attention per (b,h). 64 query rows / block.
// K tile [64 j][128 d] and V^T tile [128 d][64 j] staged via
// GLOBAL_LOAD_ASYNC_TO_LDS_B128 with padded LDS strides (no bank conflicts),
// drained with s_wait_asynccnt + barrier.
// ---------------------------------------------------------------------------
__global__ __launch_bounds__(128) void attn_kernel(
    const bf16* __restrict__ q, const bf16* __restrict__ k,
    const bf16* __restrict__ v, float* __restrict__ out) {
  __shared__ __align__(16) char smem[60672];
  bf16*  Ks   = (bf16*)smem;                   // [64][KSTR]       17408 B
  bf16*  Vt   = (bf16*)(smem + 17408);         // [128][VSTR]      18432 B
  float* Ssc  = (float*)(smem + 35840);        // [4][16][64] f32  16384 B
  bf16*  Ps   = (bf16*)(smem + 52224);         // [4][16][64] bf16  8192 B
  float* facs = (float*)(smem + 60416);        // [4][16]            256 B

  const int bh  = blockIdx.y;
  const int l0  = blockIdx.x * 64;
  const int tid = threadIdx.x, lane = tid & 31, wave = tid >> 5;
  const int kbo = (lane & 16) ? 8 : 0;
  const int mrow = lane & 15;
  const int roff = (lane & 16) ? 8 : 0;
  const bf16* qp = q + (size_t)bh * NL * NDH;   // [l][d]
  const bf16* kp = k + (size_t)bh * NL * NDH;   // [j][d]
  const bf16* vp = v + (size_t)bh * NDH * NL;   // [d][l]  (pre-transposed)
  const unsigned ksOff = (unsigned)(uintptr_t)Ks;   // LDS byte offsets
  const unsigned vtOff = (unsigned)(uintptr_t)Vt;

  // Q A-fragments: one per 32-wide K chunk of d=128
  Frag16 aq[4];
  {
    const bf16* qrow = qp + (size_t)(l0 + wave * 16 + mrow) * NDH;
#pragma unroll
    for (int c = 0; c < 4; ++c) {
      aq[c].h[0] = *(const bf16x8*)(qrow + c * 32 + kbo);
      aq[c].h[1] = *(const bf16x8*)(qrow + c * 32 + kbo + 16);
    }
  }
  v8f accO[8] = {};
  float m_run = -1e30f, l_run = 0.f;

  for (int j0 = 0; j0 < NL; j0 += 64) {
    // ---- async stage K tile: 64 rows x 128 d, padded row stride KSTR ----
#pragma unroll
    for (int it = 0; it < 8; ++it) {
      int e = it * 128 + tid;             // 0..1023 b128 chunks
      int j = e >> 4, sub = e & 15;       // 16 b128 per row
      async_ld_b128(ksOff + (unsigned)((j * KSTR + sub * 8) * 2),
                    kp + (size_t)(j0 + j) * NDH + sub * 8);
    }
    // ---- async stage V^T tile: 128 d-rows x 64 j, padded row stride VSTR ----
#pragma unroll
    for (int it = 0; it < 8; ++it) {
      int e = it * 128 + tid;             // 0..1023 b128 chunks
      int d = e >> 3, sub = e & 7;        // 8 b128 per row
      async_ld_b128(vtOff + (unsigned)((d * VSTR + sub * 8) * 2),
                    vp + (size_t)d * NL + j0 + sub * 8);
    }
    if (j0 + 64 < NL) {   // prefetch next tiles (global_prefetch_b8)
      __builtin_prefetch(kp + (size_t)(j0 + 64) * NDH + tid * 64, 0, 0);
      __builtin_prefetch(vp + (size_t)tid * NL + j0 + 64, 0, 0);
    }
    wait_async0();
    __syncthreads();

    // ---- S = Q K'^T : 16x64 per wave, K(=d)=128 via 4 chained WMMAs ----
    float* Sw = Ssc + wave * (16 * 64);
#pragma unroll
    for (int jt = 0; jt < 4; ++jt) {
      v8f s = {};
      const bf16* krow = Ks + (jt * 16 + mrow) * KSTR;  // B operand: lane = j
#pragma unroll
      for (int c = 0; c < 4; ++c) {
        Frag16 bk;
        bk.h[0] = *(const bf16x8*)(krow + c * 32 + kbo);
        bk.h[1] = *(const bf16x8*)(krow + c * 32 + kbo + 16);
        s = wmma_bf16(aq[c], bk, s);
      }
#pragma unroll
      for (int r = 0; r < 8; ++r)
        Sw[(r + roff) * 64 + jt * 16 + mrow] = s[r];
    }

    // ---- online softmax (wave-local; lane -> row mrow, half -> 32 cols) ----
    float* Srow = Sw + mrow * 64 + ((lane & 16) ? 32 : 0);
    float mloc = -1e30f;
#pragma unroll
    for (int jj2 = 0; jj2 < 32; ++jj2) mloc = fmaxf(mloc, Srow[jj2]);
    mloc = fmaxf(mloc, __shfl_xor(mloc, 16, 32));
    float mnew = fmaxf(m_run, mloc);
    float fac  = __expf(m_run - mnew);
    bf16* Prow = Ps + wave * (16 * 64) + mrow * 64 + ((lane & 16) ? 32 : 0);
    float psum = 0.f;
#pragma unroll
    for (int jj2 = 0; jj2 < 32; ++jj2) {
      float p = __expf(Srow[jj2] - mnew);
      psum += p;
      Prow[jj2] = (bf16)p;
    }
    psum += __shfl_xor(psum, 16, 32);
    l_run = l_run * fac + psum;
    m_run = mnew;
    if (lane < 16) facs[wave * 16 + lane] = fac;
    float frow[8];
#pragma unroll
    for (int r = 0; r < 8; ++r) frow[r] = facs[wave * 16 + r + roff];
#pragma unroll
    for (int dt = 0; dt < 8; ++dt)
#pragma unroll
      for (int r = 0; r < 8; ++r) accO[dt][r] *= frow[r];

    // ---- O += P @ V : A = P (K=64 keys, 2 chunks), B = V^T rows ----
    const bf16* Pm = Ps + wave * (16 * 64) + mrow * 64;
#pragma unroll
    for (int c = 0; c < 2; ++c) {
      Frag16 ap;
      ap.h[0] = *(const bf16x8*)(Pm + c * 32 + kbo);
      ap.h[1] = *(const bf16x8*)(Pm + c * 32 + kbo + 16);
#pragma unroll
      for (int dt = 0; dt < 8; ++dt) {
        Frag16 bv;
        const bf16* vrow = Vt + (dt * 16 + mrow) * VSTR;  // lane = d column
        bv.h[0] = *(const bf16x8*)(vrow + c * 32 + kbo);
        bv.h[1] = *(const bf16x8*)(vrow + c * 32 + kbo + 16);
        accO[dt] = wmma_bf16(ap, bv, accO[dt]);
      }
    }
    __syncthreads();   // guard LDS reuse next iteration
  }

  // ---- finalize: divide by row sum, transpose via LDS, coalesced store ----
  if (lane < 16) facs[wave * 16 + lane] = l_run;
  float linv[8];
#pragma unroll
  for (int r = 0; r < 8; ++r) linv[r] = 1.0f / facs[wave * 16 + r + roff];
  __syncthreads();
  float* Ot = (float*)smem;   // [128][68] f32 (reuses Ks/Vt region), 34816 B
#pragma unroll
  for (int dt = 0; dt < 8; ++dt)
#pragma unroll
    for (int r = 0; r < 8; ++r) {
      int row = wave * 16 + r + roff;
      int d = dt * 16 + mrow;
      Ot[d * 68 + row] = accO[dt][r] * linv[r];
    }
  __syncthreads();
  {
    // out[b, h*128+d, pixel] with b*INNER + h*128 + d == bh*128 + d
    int d = tid;
    float* op = out + ((size_t)(bh * 128 + d)) * NL + l0;
#pragma unroll
    for (int ll = 0; ll < 64; ll += 4)
      *(float4*)(op + ll) = *(const float4*)(Ot + d * 68 + ll);
  }
}

// ---------------------------------------------------------------------------
extern "C" void kernel_launch(void* const* d_in, const int* in_sizes, int n_in,
                              void* d_out, int out_size, void* d_ws, size_t ws_size,
                              hipStream_t stream) {
  (void)in_sizes; (void)n_in; (void)out_size; (void)ws_size;
  const float* content = (const float*)d_in[0];
  const float* style   = (const float*)d_in[1];
  const float* Wq      = (const float*)d_in[2];
  const float* Wk      = (const float*)d_in[3];
  const float* Wv      = (const float*)d_in[4];
  const float* pos_h   = (const float*)d_in[5];
  const float* pos_w   = (const float*)d_in[6];

  float* stats = (float*)d_ws;
  const size_t statsBytes = (size_t)2 * NB * NC * 2 * sizeof(float);  // 128 KiB
  const size_t qkvElems   = (size_t)NB * NHEADS * NL * NDH;           // 8.4M
  bf16* qb = (bf16*)((char*)d_ws + statsBytes);
  bf16* kb = qb + qkvElems;
  bf16* vb = kb + qkvElems;

  stats_kernel<<<dim3(2 * NB * NC), 128, 0, stream>>>(content, style, stats);
  qkv_kernel<<<dim3(NINNER / 64, NL / 64, NB * 3), 128, 0, stream>>>(
      content, style, Wq, Wk, Wv, pos_h, pos_w, stats, qb, kb, vb);
  attn_kernel<<<dim3(NL / 64, NB * NHEADS), 128, 0, stream>>>(
      qb, kb, vb, (float*)d_out);
}
